// ExpertMixture_34952443854979
// MI455X (gfx1250) — matrix operand order repaired and checked by reading, not compiled, not run on hardware
//
#include <hip/hip_runtime.h>

// Problem constants (match reference)
#define TOKENS 16384
#define DIM    1024
#define HID    2048
#define NEXP   8
#define NCLS   3
#define TILE_M 128
#define HSPLIT 4                      // H-dimension split for occupancy
#define HSLICE (HID / HSPLIT)         // 512 columns per block

typedef __attribute__((ext_vector_type(16))) __bf16 v16bf;
typedef __attribute__((ext_vector_type(8)))  float  v8f;

union Frag {
    uint4 q[2];   // 32 bytes = 8 VGPRs
    v16bf v;      // 16 bf16 elements
};

__device__ __forceinline__ unsigned short f2bf(float f) {
    unsigned int u = __float_as_uint(f);
    unsigned int r = u + 0x7FFFu + ((u >> 16) & 1u);   // round-to-nearest-even
    return (unsigned short)(r >> 16);
}

// ---------------------------------------------------------------------------
// Kernel 0: zero the per-expert counters (graph-safe, runs every call)
// ---------------------------------------------------------------------------
__global__ void init_counts_kernel(int* __restrict__ counts) {
    if (threadIdx.x < NEXP) counts[threadIdx.x] = 0;
}

// ---------------------------------------------------------------------------
// Kernel 1: router (argmax over 8 experts) + append token to expert list
//           + convert the token's x row to bf16 (for the WMMA GEMM).
// One block (256 threads) per token; each thread handles 4 of the 1024 dims.
// ---------------------------------------------------------------------------
__global__ __launch_bounds__(256)
void router_kernel(const float* __restrict__ x, const float* __restrict__ Wr,
                   const float* __restrict__ br,
                   unsigned short* __restrict__ xb,
                   int* __restrict__ counts, int* __restrict__ idx_buf)
{
    const int b   = blockIdx.x;
    const int tid = threadIdx.x;
    __shared__ float sred[8][NEXP];

    // one float4 of x per thread (256*4 == 1024)
    const float4 xv = ((const float4*)(x + (size_t)b * DIM))[tid];
    const float vals[4] = {xv.x, xv.y, xv.z, xv.w};

    float acc[NEXP];
#pragma unroll
    for (int e = 0; e < NEXP; ++e) acc[e] = 0.f;

    const int d0 = tid * 4;
#pragma unroll
    for (int j = 0; j < 4; ++j) {
        const float* wr = Wr + (size_t)(d0 + j) * NEXP;
        const float4 w0 = ((const float4*)wr)[0];
        const float4 w1 = ((const float4*)wr)[1];
        const float xd = vals[j];
        acc[0] += xd * w0.x; acc[1] += xd * w0.y;
        acc[2] += xd * w0.z; acc[3] += xd * w0.w;
        acc[4] += xd * w1.x; acc[5] += xd * w1.y;
        acc[6] += xd * w1.z; acc[7] += xd * w1.w;
    }

    // store bf16 copy of x row (4 elems -> 8 bytes)
    uint2 pk;
    pk.x = (unsigned int)f2bf(xv.x) | ((unsigned int)f2bf(xv.y) << 16);
    pk.y = (unsigned int)f2bf(xv.z) | ((unsigned int)f2bf(xv.w) << 16);
    ((uint2*)xb)[(size_t)b * (DIM / 4) + tid] = pk;

    // wave32 reduction
#pragma unroll
    for (int off = 16; off > 0; off >>= 1) {
#pragma unroll
        for (int e = 0; e < NEXP; ++e) acc[e] += __shfl_xor(acc[e], off, 32);
    }
    const int wave = tid >> 5;
    if ((tid & 31) == 0) {
#pragma unroll
        for (int e = 0; e < NEXP; ++e) sred[wave][e] = acc[e];
    }
    __syncthreads();

    if (tid == 0) {
        float best = -3.4e38f;
        int   bestE = 0;
#pragma unroll
        for (int e = 0; e < NEXP; ++e) {
            float tot = br[e];
#pragma unroll
            for (int w = 0; w < 8; ++w) tot += sred[w][e];
            if (tot > best) { best = tot; bestE = e; }
        }
        const int pos = atomicAdd(&counts[bestE], 1);
        idx_buf[bestE * TOKENS + pos] = b;
    }
}

// ---------------------------------------------------------------------------
// Kernel 2: W1 [E, D, H] fp32  ->  W1t [E, H, D] bf16  (LDS-tiled transpose)
// grid: (H/32, D/32, E), block: (32, 8)
// ---------------------------------------------------------------------------
__global__ __launch_bounds__(256)
void w1_transpose_kernel(const float* __restrict__ W1,
                         unsigned short* __restrict__ w1t)
{
    __shared__ float tile[32][33];
    const int e  = blockIdx.z;
    const int h0 = blockIdx.x * 32;
    const int d0 = blockIdx.y * 32;
    const int tx = threadIdx.x, ty = threadIdx.y;

    const float* src = W1 + ((size_t)e * DIM + d0) * HID + h0;
#pragma unroll
    for (int k = 0; k < 4; ++k) {
        const int i = ty + k * 8;                  // d offset within tile
        tile[i][tx] = src[(size_t)i * HID + tx];   // coalesced over h
    }
    __syncthreads();

    unsigned short* dst = w1t + ((size_t)e * HID + h0) * DIM + d0;
#pragma unroll
    for (int k = 0; k < 4; ++k) {
        const int i = ty + k * 8;                          // h offset
        dst[(size_t)i * DIM + tx] = f2bf(tile[tx][i]);     // coalesced over d
    }
}

// ---------------------------------------------------------------------------
// Kernel 3: grouped GEMM per expert with fused bias+ReLU+head.
// grid = (136, HSPLIT): x = 128-token tile of one expert, y = 512-wide H
// slice. 8 waves/block; wave w owns rows [16w,16w+16); per 64-wide H chunk
// it keeps 4 f32 accumulators and runs v_wmma_f32_16x16x32_bf16 over K=1024.
// Epilogue folds relu(h+b1) @ W2 into per-lane partials, reduced once; each
// block writes its H-slice partial (b2 folded into slice 0) to workspace.
// ---------------------------------------------------------------------------
__global__ __launch_bounds__(256)
void moe_gemm_kernel(const unsigned short* __restrict__ xb,
                     const unsigned short* __restrict__ w1t,
                     const float* __restrict__ b1,
                     const float* __restrict__ W2,
                     const float* __restrict__ b2,
                     const int* __restrict__ counts,
                     const int* __restrict__ idx_buf,
                     float* __restrict__ pout)
{
    __shared__ int   sIdx[TILE_M];
    __shared__ float sW2[HSLICE * NCLS];   // 6 KB
    __shared__ float sb1[HSLICE];          // 2 KB

    // ---- map blockIdx.x -> (expert, tile) via prefix over ceil(count/128) --
    const int bid   = blockIdx.x;
    const int split = blockIdx.y;
    int e = -1, t = 0, run = 0;
    for (int i = 0; i < NEXP; ++i) {
        const int tiles = (counts[i] + TILE_M - 1) / TILE_M;
        if (e < 0 && bid < run + tiles) { e = i; t = bid - run; }
        run += tiles;
    }
    if (e < 0) return;                       // uniform per block
    const int cnt     = counts[e];
    const int m_valid = min(TILE_M, cnt - t * TILE_M);
    const int h_base  = split * HSLICE;

    const int tid = threadIdx.x;
    if (tid < TILE_M) {
        const int src = t * TILE_M + ((tid < m_valid) ? tid : 0); // pad w/ row 0
        sIdx[tid] = idx_buf[e * TOKENS + src];
    }
    for (int i = tid; i < HSLICE * NCLS; i += 256)
        sW2[i] = W2[(size_t)e * HID * NCLS + (size_t)h_base * NCLS + i];
    for (int i = tid; i < HSLICE; i += 256)
        sb1[i] = b1[(size_t)e * HID + h_base + i];
    __syncthreads();

    const int wave   = tid >> 5;
    const int lane   = tid & 31;
    const int nl     = lane & 15;    // A row / B col / C col within fragment
    const int half   = lane >> 4;    // K-half selector, C row-group selector
    const int m_base = wave * 16;

    const unsigned short* arow  = xb + (size_t)sIdx[m_base + nl] * DIM;
    const unsigned short* wbase = w1t + (size_t)e * HID * DIM;

    float lane_out[8][NCLS] = {};

    for (int hc = 0; hc < HSLICE / 64; ++hc) {     // 8 chunks of 64
        v8f accv[4];
#pragma unroll
        for (int ni = 0; ni < 4; ++ni)
#pragma unroll
            for (int j = 0; j < 8; ++j) accv[ni][j] = 0.f;

        const unsigned short* brow[4];
#pragma unroll
        for (int ni = 0; ni < 4; ++ni)
            brow[ni] = wbase + (size_t)(h_base + hc * 64 + ni * 16 + nl) * DIM;

        if (hc + 1 < HSLICE / 64)  // hint next H-chunk of W1t (global_prefetch_b8)
            __builtin_prefetch(wbase + (size_t)(h_base + (hc + 1) * 64 + nl) * DIM + half * 16, 0, 0);

#pragma unroll 2
        for (int kc = 0; kc < DIM; kc += 32) {
            // A fragment (16x32 bf16): VGPR0-3 K = 8*half..8*half+7,
            //                          VGPR4-7 K = 16+8*half..
            Frag a;
            a.q[0] = *(const uint4*)(arow + kc + half * 8);
            a.q[1] = *(const uint4*)(arow + kc + 16 + half * 8);
            const int kb = kc + half * 8;   // B fragment: same K order as A
#pragma unroll
            for (int ni = 0; ni < 4; ++ni) {
                Frag bf;
                bf.q[0] = *(const uint4*)(brow[ni] + kb);
                bf.q[1] = *(const uint4*)(brow[ni] + kb + 16);
                accv[ni] = __builtin_amdgcn_wmma_f32_16x16x32_bf16(
                    false, a.v, false, bf.v, (short)0, accv[ni], false, false);
            }
        }

        // fused epilogue: h = relu(acc + b1); partial_out += h * W2
#pragma unroll
        for (int ni = 0; ni < 4; ++ni) {
            const int h_loc = hc * 64 + ni * 16 + nl;   // column owned by lane
            const float bias = sb1[h_loc];
            const float w2c0 = sW2[h_loc * NCLS + 0];
            const float w2c1 = sW2[h_loc * NCLS + 1];
            const float w2c2 = sW2[h_loc * NCLS + 2];
#pragma unroll
            for (int r = 0; r < 8; ++r) {
                float v = accv[ni][r] + bias;
                v = v > 0.f ? v : 0.f;
                lane_out[r][0] += v * w2c0;
                lane_out[r][1] += v * w2c1;
                lane_out[r][2] += v * w2c2;
            }
        }
    }

    // reduce the 16 column-lanes within each half (rows differ by `half`)
#pragma unroll
    for (int r = 0; r < 8; ++r)
#pragma unroll
        for (int c = 0; c < NCLS; ++c) {
            float v = lane_out[r][c];
            v += __shfl_xor(v, 1, 32);
            v += __shfl_xor(v, 2, 32);
            v += __shfl_xor(v, 4, 32);
            v += __shfl_xor(v, 8, 32);
            lane_out[r][c] = v;
        }

    if (nl == 0) {
        const float a0 = (split == 0) ? b2[e * NCLS + 0] : 0.f;
        const float a1 = (split == 0) ? b2[e * NCLS + 1] : 0.f;
        const float a2 = (split == 0) ? b2[e * NCLS + 2] : 0.f;
        float* pbase = pout + (size_t)split * TOKENS * NCLS;
#pragma unroll
        for (int r = 0; r < 8; ++r) {
            const int m = m_base + half * 8 + r;   // C layout: row = 8*half + vgpr
            if (m < m_valid) {
                const int token = sIdx[m];
                pbase[(size_t)token * NCLS + 0] = lane_out[r][0] + a0;
                pbase[(size_t)token * NCLS + 1] = lane_out[r][1] + a1;
                pbase[(size_t)token * NCLS + 2] = lane_out[r][2] + a2;
            }
        }
    }
}

// ---------------------------------------------------------------------------
// Kernel 4: deterministic combine of the HSPLIT partial outputs.
// ---------------------------------------------------------------------------
__global__ __launch_bounds__(256)
void combine_kernel(const float* __restrict__ pout, float* __restrict__ out)
{
    const int i = blockIdx.x * 256 + threadIdx.x;   // over TOKENS*NCLS
    if (i < TOKENS * NCLS) {
        float s = 0.f;
#pragma unroll
        for (int sp = 0; sp < HSPLIT; ++sp)
            s += pout[(size_t)sp * TOKENS * NCLS + i];
        out[i] = s;
    }
}

// ---------------------------------------------------------------------------
extern "C" void kernel_launch(void* const* d_in, const int* in_sizes, int n_in,
                              void* d_out, int out_size, void* d_ws, size_t ws_size,
                              hipStream_t stream) {
    const float* x  = (const float*)d_in[0];   // [B, D]
    const float* Wr = (const float*)d_in[1];   // [D, E]
    const float* br = (const float*)d_in[2];   // [E]
    const float* W1 = (const float*)d_in[3];   // [E, D, H]
    const float* b1 = (const float*)d_in[4];   // [E, H]
    const float* W2 = (const float*)d_in[5];   // [E, H, C]
    const float* b2 = (const float*)d_in[6];   // [E, C]
    float* out = (float*)d_out;                // [B, C]

    // workspace layout
    char* ws = (char*)d_ws;
    int* counts  = (int*)ws;                               // 8 ints (pad 256B)
    int* idx_buf = (int*)(ws + 256);                       // E*B ints = 512 KB
    unsigned short* xb  = (unsigned short*)(ws + 256 + (size_t)NEXP * TOKENS * 4);
    unsigned short* w1t = xb + (size_t)TOKENS * DIM;       // each 32 MB
    float* pout = (float*)(w1t + (size_t)NEXP * HID * DIM); // HSPLIT*B*C floats

    init_counts_kernel<<<1, 32, 0, stream>>>(counts);
    router_kernel<<<TOKENS, 256, 0, stream>>>(x, Wr, br, xb, counts, idx_buf);
    w1_transpose_kernel<<<dim3(HID / 32, DIM / 32, NEXP), dim3(32, 8), 0, stream>>>(W1, w1t);

    const int n_tiles = TOKENS / TILE_M + NEXP;            // upper bound = 136
    moe_gemm_kernel<<<dim3(n_tiles, HSPLIT), 256, 0, stream>>>(
        xb, w1t, b1, W2, b2, counts, idx_buf, pout);

    combine_kernel<<<(TOKENS * NCLS + 255) / 256, 256, 0, stream>>>(pout, out);
}
